// ImageTransformerEncoder_3152505996016
// MI455X (gfx1250) — matrix operand order repaired and checked by reading, not compiled, not run on hardware
//
#include <hip/hip_runtime.h>
#include <math.h>

// ---------------------------------------------------------------------------
// Types for CDNA5 WMMA (wave32): v_wmma_f32_16x16x32_bf16
// ---------------------------------------------------------------------------
typedef __attribute__((ext_vector_type(16))) __bf16 v16bf;
typedef __attribute__((ext_vector_type(8)))  __bf16 v8bf;
typedef __attribute__((ext_vector_type(8)))  float  v8f;

#define S_LEN   2304   // 48*48
#define HID     256
#define NH      8
#define QKN     2048   // HID*NH
#define MLP     1024   // 4*HID

// ---------------------------------------------------------------------------
// Fragment load helpers (two ds_load_b128 each).
// A-fragment (16x32 bf16): lane m = lane&15, half = lane>>4.
//   elems 0..7  -> k =  8*half + 0..7
//   elems 8..15 -> k = 16 + 8*half + 0..7
// B-fragment (32x16 bf16), stored [n][k] in LDS: lane n = lane&15, half=lane>>4.
//   elems 0..7  -> k =  8*half + 0..7, elems 8..15 -> k = 16 + 8*half + 0..7
// ---------------------------------------------------------------------------
__device__ __forceinline__ v16bf ld_frag(const __bf16* p0, const __bf16* p1) {
    v8bf lo = *(const v8bf*)p0;
    v8bf hi = *(const v8bf*)p1;
    v16bf r;
#pragma unroll
    for (int i = 0; i < 8; ++i) { r[i] = lo[i]; r[i + 8] = hi[i]; }
    return r;
}

__device__ __forceinline__ v8f wmma_bf16(v16bf a, v16bf b, v8f c) {
    return __builtin_amdgcn_wmma_f32_16x16x32_bf16(
        false, a, false, b, (short)0, c, false, false);
}

// f32 -> bf16 packers (8 at a time -> one 16B LDS store)
__device__ __forceinline__ v8bf cvt8_f4(float4 a, float4 b) {
    v8bf r;
    r[0] = (__bf16)a.x; r[1] = (__bf16)a.y; r[2] = (__bf16)a.z; r[3] = (__bf16)a.w;
    r[4] = (__bf16)b.x; r[5] = (__bf16)b.y; r[6] = (__bf16)b.z; r[7] = (__bf16)b.w;
    return r;
}
__device__ __forceinline__ v8bf cvt8_arr(const float* f) {
    v8bf r;
#pragma unroll
    for (int i = 0; i < 8; ++i) r[i] = (__bf16)f[i];
    return r;
}

// ---------------------------------------------------------------------------
// Embedding: out[s][c] = sin(wk*i) + cos(wk*j) + sum_t x[t][s]*semW[t][c] + semb[c]
// ---------------------------------------------------------------------------
__global__ __launch_bounds__(256) void embed_kernel(
    const float* __restrict__ x, const float* __restrict__ semW,
    const float* __restrict__ semb, float* __restrict__ out) {
    int s = blockIdx.x;
    int c = threadIdx.x;
    float fi = (float)(s / 48);
    float fj = (float)(s % 48);
    float wk = __expf((float)c * (-9.210340371976184f / 256.0f)); // 10000^(-c/256)
    float acc = __sinf(wk * fi) + __cosf(wk * fj) + semb[c];
#pragma unroll
    for (int t = 0; t < 3; ++t)
        acc += x[t * S_LEN + s] * semW[t * HID + c];
    out[s * HID + c] = acc;
}

// ---------------------------------------------------------------------------
// GEMM: C[M,N] = epilogue(A[M,K] @ B[K,N]); f32 in memory, bf16 in LDS,
// WMMA f32 accumulate. 128x128 block tile, 8 waves x (64x32), BK=32.
// Double-buffered LDS: global loads for chunk i+1 overlap WMMAs on chunk i.
// Staging: A via global_load_b128, B via strided scalar loads; ds_store_b128.
// Epilogue: v = (acc + bias[n]) * scale; [ReLU]; [+= res[m][n]]
// ---------------------------------------------------------------------------
template <bool RELU, bool HAS_RES>
__global__ __launch_bounds__(256) void gemm_bf16_kernel(
    const float* __restrict__ A, const float* __restrict__ B,
    const float* __restrict__ bias, const float* __restrict__ res,
    float* __restrict__ C, int M, int N, int K, float scale) {
    __shared__ __align__(16) __bf16 Alds[2][128 * 32];   // [m][k]
    __shared__ __align__(16) __bf16 Blds[2][128 * 32];   // [n][k]

    const int tid   = threadIdx.x;
    const int wave  = tid >> 5;
    const int lane  = tid & 31;
    const int hf    = lane >> 4;
    const int lm    = lane & 15;
    const int waveM = wave & 1;    // 0..1 -> 64 rows
    const int waveN = wave >> 1;   // 0..3 -> 32 cols
    const int m0 = blockIdx.y * 128;
    const int n0 = blockIdx.x * 128;

    v8f acc[4][2];
#pragma unroll
    for (int i = 0; i < 4; ++i)
#pragma unroll
        for (int j = 0; j < 2; ++j) acc[i][j] = (v8f)(0.0f);

    // Register staging: A = 2 runs of 8 contiguous k; B = 2 runs of 8 k
    // gathered down a column (coalesced across lanes).
    float4 aR[4];
    float  bR[2][8];

    auto load_regs = [&](int k0) {
#pragma unroll
        for (int j = 0; j < 2; ++j) {
            int u = tid + j * 256;             // 0..511
            int r = u >> 2, k8 = (u & 3) * 8;  // 128 rows x 4 octets
            const float* p = A + (m0 + r) * K + k0 + k8;
            aR[2 * j]     = *(const float4*)p;
            aR[2 * j + 1] = *(const float4*)(p + 4);
        }
#pragma unroll
        for (int j = 0; j < 2; ++j) {
            int u = tid + j * 256;
            int n = u & 127, ko = u >> 7;      // 128 cols x 4 k-octets
            const float* p = B + (k0 + ko * 8) * N + n0 + n;
#pragma unroll
            for (int i = 0; i < 8; ++i) bR[j][i] = p[i * N];
        }
    };
    auto store_lds = [&](int buf) {
#pragma unroll
        for (int j = 0; j < 2; ++j) {
            int u = tid + j * 256;
            int r = u >> 2, k8 = (u & 3) * 8;
            *(v8bf*)&Alds[buf][r * 32 + k8] = cvt8_f4(aR[2 * j], aR[2 * j + 1]);
        }
#pragma unroll
        for (int j = 0; j < 2; ++j) {
            int u = tid + j * 256;
            int n = u & 127, ko = u >> 7;
            *(v8bf*)&Blds[buf][n * 32 + ko * 8] = cvt8_arr(bR[j]);
        }
    };

    const int nCh = K >> 5;
    load_regs(0);
    store_lds(0);

    for (int i = 0; i < nCh; ++i) {
        __syncthreads();
        if (i + 1 < nCh) load_regs((i + 1) << 5);   // overlap with WMMAs

        const __bf16* Ab = Alds[i & 1];
        const __bf16* Bb = Blds[i & 1];
        v16bf afr[4];
#pragma unroll
        for (int tm = 0; tm < 4; ++tm) {
            int row = waveM * 64 + tm * 16 + lm;
            afr[tm] = ld_frag(&Ab[row * 32 + 8 * hf],
                              &Ab[row * 32 + 16 + 8 * hf]);
        }
#pragma unroll
        for (int tn = 0; tn < 2; ++tn) {
            int col = waveN * 32 + tn * 16 + lm;
            v16bf bfr = ld_frag(&Bb[col * 32 + 8 * hf],
                                &Bb[col * 32 + 16 + 8 * hf]);
#pragma unroll
            for (int tm = 0; tm < 4; ++tm)
                acc[tm][tn] = wmma_bf16(afr[tm], bfr, acc[tm][tn]);
        }

        if (i + 1 < nCh) store_lds((i + 1) & 1);
    }

    // Epilogue. D layout: elem r -> m = r + 8*half, n = lane&15.
#pragma unroll
    for (int tn = 0; tn < 2; ++tn) {
        int n = n0 + waveN * 32 + tn * 16 + lm;
        float bv = bias ? bias[n] : 0.0f;
#pragma unroll
        for (int tm = 0; tm < 4; ++tm) {
#pragma unroll
            for (int r = 0; r < 8; ++r) {
                int m = m0 + waveM * 64 + tm * 16 + r + 8 * hf;
                float v = (acc[tm][tn][r] + bv) * scale;
                if (RELU) v = fmaxf(v, 0.0f);
                if (HAS_RES) v += res[m * N + n];
                C[m * N + n] = v;
            }
        }
    }
}

// ---------------------------------------------------------------------------
// Flash attention (per head). Q,K,V f32 (S_LEN x 2048); head h = cols
// h*256..h*256+255. Scale 0.25 folded into Q and K upstream.
// Grid: (18 q-blocks of 128 rows, 8 heads). 8 waves, each owns 16 q rows.
// ---------------------------------------------------------------------------
__global__ __launch_bounds__(256) void flash_attn_kernel(
    const float* __restrict__ Q, const float* __restrict__ K,
    const float* __restrict__ V, float* __restrict__ O) {
    __shared__ __align__(16) __bf16 Qs[128 * 256];   // [r][d]  64 KB
    __shared__ __align__(16) __bf16 Ks[32 * 256];    // [j][d]  16 KB
    __shared__ __align__(16) __bf16 Vs[256 * 32];    // [d][j]  16 KB (transposed)
    __shared__ __align__(16) __bf16 Ps[8 * 16 * 32]; // per-wave P scratch, 8 KB

    const int tid  = threadIdx.x;
    const int w    = tid >> 5;
    const int lane = tid & 31;
    const int hf   = lane >> 4;
    const int lm   = lane & 15;
    const int qb   = blockIdx.x;
    const int head = blockIdx.y;
    const int hbase = head * HID;

    // Stage Q block once: b128 loads + b128 DS stores.
#pragma unroll 4
    for (int i = 0; i < 16; ++i) {
        int u = tid + i * 256;                 // 0..4095 runs of 8
        int r = u >> 5, d8 = (u & 31) * 8;
        const float* p = Q + (qb * 128 + r) * QKN + hbase + d8;
        *(v8bf*)&Qs[r * 256 + d8] = cvt8_f4(*(const float4*)p, *(const float4*)(p + 4));
    }

    v8f Oacc[16];
#pragma unroll
    for (int t = 0; t < 16; ++t) Oacc[t] = (v8f)(0.0f);
    float rm[8], rl[8];
#pragma unroll
    for (int r = 0; r < 8; ++r) { rm[r] = -INFINITY; rl[r] = 0.0f; }

    __bf16* Pw = &Ps[w * 512];

    for (int jb = 0; jb < S_LEN; jb += 32) {
        __syncthreads();
        // K rows: contiguous b128; V: gather 8 j-strided (coalesced over d),
        // store transposed [d][j] as b128.
#pragma unroll
        for (int i = 0; i < 4; ++i) {
            int u = tid + i * 256;             // 0..1023
            int j = u >> 5, d8 = (u & 31) * 8;
            const float* p = K + (jb + j) * QKN + hbase + d8;
            *(v8bf*)&Ks[j * 256 + d8] = cvt8_f4(*(const float4*)p, *(const float4*)(p + 4));
        }
#pragma unroll
        for (int i = 0; i < 4; ++i) {
            int u = tid + i * 256;
            int d = u & 255, jo = u >> 8;      // 256 d x 4 j-octets
            const float* p = V + (jb + jo * 8) * QKN + hbase + d;
            float t[8];
#pragma unroll
            for (int q = 0; q < 8; ++q) t[q] = p[q * QKN];
            *(v8bf*)&Vs[d * 32 + jo * 8] = cvt8_arr(t);
        }
        __syncthreads();

        // S = Qw(16x256) @ K^T(256x32) -> two 16x16 f32 tiles
        v8f s0 = (v8f)(0.0f), s1 = (v8f)(0.0f);
#pragma unroll
        for (int c = 0; c < 8; ++c) {
            const __bf16* qrow = &Qs[(w * 16 + lm) * 256 + c * 32];
            v16bf aq = ld_frag(qrow + 8 * hf, qrow + 16 + 8 * hf);
            const __bf16* k0 = &Ks[lm * 256 + c * 32 + 8 * hf];
            const __bf16* k1 = &Ks[(16 + lm) * 256 + c * 32 + 8 * hf];
            v16bf b0 = ld_frag(k0, k0 + 16);
            v16bf b1 = ld_frag(k1, k1 + 16);
            s0 = wmma_bf16(aq, b0, s0);
            s1 = wmma_bf16(aq, b1, s1);
        }

        // Online softmax per row (row m = r + 8*half lives in one 16-lane half).
        float corr[8];
#pragma unroll
        for (int r = 0; r < 8; ++r) {
            float mx = fmaxf(s0[r], s1[r]);
#pragma unroll
            for (int msk = 1; msk < 16; msk <<= 1)
                mx = fmaxf(mx, __shfl_xor(mx, msk, 32));
            float mn = fmaxf(rm[r], mx);
            corr[r] = __expf(rm[r] - mn);
            float p0 = __expf(s0[r] - mn);
            float p1 = __expf(s1[r] - mn);
            float rs = p0 + p1;
#pragma unroll
            for (int msk = 1; msk < 16; msk <<= 1)
                rs += __shfl_xor(rs, msk, 32);
            rl[r] = rl[r] * corr[r] + rs;
            rm[r] = mn;
            Pw[(r + 8 * hf) * 32 + lm]      = (__bf16)p0;
            Pw[(r + 8 * hf) * 32 + 16 + lm] = (__bf16)p1;
        }
#pragma unroll
        for (int t = 0; t < 16; ++t)
#pragma unroll
            for (int r = 0; r < 8; ++r) Oacc[t][r] *= corr[r];

        // O += P(16x32) @ V(32x256)
        v16bf pf = ld_frag(&Pw[lm * 32 + 8 * hf], &Pw[lm * 32 + 16 + 8 * hf]);
#pragma unroll
        for (int t = 0; t < 16; ++t) {
            const __bf16* vp = &Vs[(t * 16 + lm) * 32 + 8 * hf];
            v16bf vf = ld_frag(vp, vp + 16);
            Oacc[t] = wmma_bf16(pf, vf, Oacc[t]);
        }
    }

    // Normalize and write out.
#pragma unroll
    for (int r = 0; r < 8; ++r) {
        float inv = 1.0f / rl[r];
        int s = qb * 128 + w * 16 + r + 8 * hf;
#pragma unroll
        for (int t = 0; t < 16; ++t)
            O[s * QKN + hbase + t * 16 + lm] = Oacc[t][r] * inv;
    }
}

// ---------------------------------------------------------------------------
// Row LayerNorm over 256 features. One row per block.
// ---------------------------------------------------------------------------
__global__ __launch_bounds__(256) void layernorm_kernel(
    const float* __restrict__ in, const float* __restrict__ g,
    const float* __restrict__ b, float* __restrict__ out) {
    __shared__ float red[256];
    int row = blockIdx.x, tid = threadIdx.x;
    float v = in[row * HID + tid];
    red[tid] = v;
    __syncthreads();
#pragma unroll
    for (int s = 128; s > 0; s >>= 1) {
        if (tid < s) red[tid] += red[tid + s];
        __syncthreads();
    }
    float mean = red[0] * (1.0f / HID);
    __syncthreads();
    float d = v - mean;
    red[tid] = d * d;
    __syncthreads();
#pragma unroll
    for (int s = 128; s > 0; s >>= 1) {
        if (tid < s) red[tid] += red[tid + s];
        __syncthreads();
    }
    float var = red[0] * (1.0f / HID);
    out[row * HID + tid] = g[tid] * d * rsqrtf(var + 1e-5f) + b[tid];
}

// ---------------------------------------------------------------------------
// Final transpose: (S, 256) -> (256, S)
// ---------------------------------------------------------------------------
__global__ __launch_bounds__(256) void transpose_kernel(
    const float* __restrict__ in, float* __restrict__ out) {
    int s = blockIdx.x, c = threadIdx.x;
    out[c * S_LEN + s] = in[s * HID + c];
}

// ---------------------------------------------------------------------------
// Host-side orchestration
// ---------------------------------------------------------------------------
extern "C" void kernel_launch(void* const* d_in, const int* in_sizes, int n_in,
                              void* d_out, int out_size, void* d_ws, size_t ws_size,
                              hipStream_t stream) {
    const float* x    = (const float*)d_in[0];
    const float* semW = (const float*)d_in[1];
    const float* semb = (const float*)d_in[2];
    const float* Wq   = (const float*)d_in[3];
    const float* bq   = (const float*)d_in[4];
    const float* Wk   = (const float*)d_in[5];
    const float* bk   = (const float*)d_in[6];
    const float* Wv   = (const float*)d_in[7];
    const float* bv   = (const float*)d_in[8];
    const float* Wu   = (const float*)d_in[9];
    const float* bu   = (const float*)d_in[10];
    const float* W1   = (const float*)d_in[11];
    const float* b1   = (const float*)d_in[12];
    const float* W2   = (const float*)d_in[13];
    const float* b2   = (const float*)d_in[14];
    const float* ln1g = (const float*)d_in[15];
    const float* ln1b = (const float*)d_in[16];
    const float* ln2g = (const float*)d_in[17];
    const float* ln2b = (const float*)d_in[18];

    float* ws = (float*)d_ws;
    const size_t SC  = (size_t)S_LEN * HID;   // 589824
    const size_t SQ  = (size_t)S_LEN * QKN;   // 4718592
    float* OUT   = ws;                // residual stream (S, 256)
    float* IDENT = OUT + SC;
    float* TMP   = IDENT + SC;
    float* QB    = TMP + SC;
    float* KB    = QB + SQ;
    float* VB    = KB + SQ;
    float* OB    = VB + SQ;
    float* H1    = OB + SQ;           // (S, 1024)

    const float qk_scale = 0.25f;     // 256^(-1/4)

    embed_kernel<<<S_LEN, 256, 0, stream>>>(x, semW, semb, OUT);

    for (int l = 0; l < 2; ++l) {
        const float* Wq_l = Wq + (size_t)l * HID * QKN;
        const float* Wk_l = Wk + (size_t)l * HID * QKN;
        const float* Wv_l = Wv + (size_t)l * HID * QKN;
        const float* Wu_l = Wu + (size_t)l * QKN * HID;
        const float* W1_l = W1 + (size_t)l * HID * MLP;
        const float* W2_l = W2 + (size_t)l * MLP * HID;

        dim3 gQKV(QKN / 128, S_LEN / 128);
        gemm_bf16_kernel<false, false><<<gQKV, 256, 0, stream>>>(
            OUT, Wq_l, bq + l * QKN, nullptr, QB, S_LEN, QKN, HID, qk_scale);
        gemm_bf16_kernel<false, false><<<gQKV, 256, 0, stream>>>(
            OUT, Wk_l, bk + l * QKN, nullptr, KB, S_LEN, QKN, HID, qk_scale);
        gemm_bf16_kernel<false, false><<<gQKV, 256, 0, stream>>>(
            OUT, Wv_l, bv + l * QKN, nullptr, VB, S_LEN, QKN, HID, 1.0f);

        flash_attn_kernel<<<dim3(S_LEN / 128, NH), 256, 0, stream>>>(QB, KB, VB, OB);

        gemm_bf16_kernel<false, true><<<dim3(HID / 128, S_LEN / 128), 256, 0, stream>>>(
            OB, Wu_l, bu + l * HID, OUT, TMP, S_LEN, HID, QKN, 1.0f);
        layernorm_kernel<<<S_LEN, 256, 0, stream>>>(TMP, ln1g + l * HID, ln1b + l * HID, IDENT);

        gemm_bf16_kernel<true, false><<<dim3(MLP / 128, S_LEN / 128), 256, 0, stream>>>(
            IDENT, W1_l, b1 + l * MLP, nullptr, H1, S_LEN, MLP, HID, 1.0f);
        gemm_bf16_kernel<false, true><<<dim3(HID / 128, S_LEN / 128), 256, 0, stream>>>(
            H1, W2_l, b2 + l * HID, IDENT, TMP, S_LEN, HID, MLP, 1.0f);
        layernorm_kernel<<<S_LEN, 256, 0, stream>>>(TMP, ln2g + l * HID, ln2b + l * HID, OUT);
    }

    transpose_kernel<<<S_LEN, 256, 0, stream>>>(OUT, (float*)d_out);
}
